// MyLayer_12180527251595
// MI455X (gfx1250) — matrix operand order repaired and checked by reading, not compiled, not run on hardware
//
#include <hip/hip_runtime.h>
#include <hip/hip_bf16.h>

typedef unsigned short u16;
typedef __attribute__((ext_vector_type(8)))  unsigned short v8us;
typedef __attribute__((ext_vector_type(16))) unsigned short v16us;
typedef __attribute__((ext_vector_type(16))) __bf16        v16bf;
typedef __attribute__((ext_vector_type(8)))  float         v8f;

#define MID 128
#define NT  8            // 128 / 16 n-tiles
#define KS1 5            // ceil(144/32) k-steps for GEMM1 (K padded to 160)
#define KS2 4            // 128/32 k-steps
#define WAVES 8
#define TPW_E 8          // 32-edge tiles per wave
#define TPW_N 4          // 16-node tiles per wave
#define STRIDE_E 160     // staged row width (bf16), K=144 padded to 160
#define STRIDE_N 128

static __device__ __forceinline__ u16 f2bf(float f) {
    unsigned u = __float_as_uint(f);
    unsigned r = u + 0x7FFFu + ((u >> 16) & 1u);   // round-to-nearest-even
    return (u16)(r >> 16);
}

static __device__ __forceinline__ v16bf make_frag(v8us lo, v8us hi) {
    v16us w = __builtin_shufflevector(lo, hi, 0,1,2,3,4,5,6,7,8,9,10,11,12,13,14,15);
    return __builtin_bit_cast(v16bf, w);
}

// ---------------------------------------------------------------- zero scratch
__global__ void zero_kernel(float* __restrict__ p, long n) {
    long i = (long)blockIdx.x * blockDim.x + threadIdx.x;
    if (i < n) p[i] = 0.0f;
}

// --------------------------------------------------- pack weights, B-frag bf16
// B (KxN=128) fragment layout per ISA: lane&15 = column within n-tile,
// lane>>4 selects the 8-K group; element j -> k = ks*32 + (j>>3)*16 + (l>>4)*8 + (j&7).
__global__ void pack_w_kernel(const float* __restrict__ W, u16* __restrict__ dst,
                              int Kreal, int ksteps) {
    int total = NT * ksteps * 32 * 16;
    int idx = blockIdx.x * blockDim.x + threadIdx.x;
    if (idx >= total) return;
    int j  = idx & 15;
    int l  = (idx >> 4) & 31;
    int ks = (idx >> 9) % ksteps;
    int nt = idx / (ksteps * 512);
    int col = nt * 16 + (l & 15);
    int k   = ks * 32 + (j >> 3) * 16 + (l >> 4) * 8 + (j & 7);
    float v = (k < Kreal) ? W[k * MID + col] : 0.0f;
    dst[idx] = f2bf(v);
}

// -------------------------------------------------------------- edge MLP + scatter
// Each wave owns a 32-edge tile (M=32): two A fragments per k-step reuse each
// B fragment twice, halving LDS B traffic per FLOP.
__global__ __launch_bounds__(256)
void edge_kernel(const float* __restrict__ x, const int* __restrict__ ei,
                 const float* __restrict__ ea,
                 const float* __restrict__ b1, const float* __restrict__ b2,
                 const u16* __restrict__ gW1, const u16* __restrict__ gW2,
                 float* __restrict__ summed, float* __restrict__ cnt, int E) {
    __shared__ alignas(16) u16 sW1[NT * KS1 * 32 * 16];          // 40 KB
    __shared__ alignas(16) u16 sW2[NT * KS2 * 32 * 16];          // 32 KB
    __shared__ alignas(16) u16 sStage[WAVES][32 * STRIDE_E];     // 80 KB

    const int tid = threadIdx.x, w = tid >> 5, lane = tid & 31;

    { // cooperative weight load (uint4 = 8 bf16 each)
        const uint4* g1 = (const uint4*)gW1; uint4* s1 = (uint4*)sW1;
        for (int i = tid; i < (NT * KS1 * 32 * 16) / 8; i += 256) s1[i] = g1[i];
        const uint4* g2 = (const uint4*)gW2; uint4* s2 = (uint4*)sW2;
        for (int i = tid; i < (NT * KS2 * 32 * 16) / 8; i += 256) s2[i] = g2[i];
    }
    __syncthreads();

    const int* rowI = ei;            // destination nodes
    const int* colI = ei + E;        // source nodes
    const float4* x4  = (const float4*)x;
    const float4* ea4 = (const float4*)ea;
    const int nTiles = (E + 31) / 32;
    const int tile0  = ((int)blockIdx.x * WAVES + w) * TPW_E;
    u16* st = &sStage[w][0];

    const int rA = lane & 15, kh = (lane >> 4) * 8;   // A-frag addressing
    const int n0 = lane & 15, ms = (lane >> 4) * 8;   // C/D-frag addressing
    const v8f zv = {0.f,0.f,0.f,0.f,0.f,0.f,0.f,0.f};

    // loop-invariant per-lane biases
    float b1v[NT], b2v[NT];
    #pragma unroll
    for (int nt = 0; nt < NT; ++nt) { b1v[nt] = b1[nt * 16 + n0]; b2v[nt] = b2[nt * 16 + n0]; }

    for (int t = 0; t < TPW_E; ++t) {
        const int tile = tile0 + t;
        const int base = tile * 32;
        const bool valid = (tile < nTiles);

        if (valid) { // stage 32 gathered rows + edge attrs as bf16 (A layout, row-major [m][k])
            int e = base + lane; if (e >= E) e = E - 1;
            const int c = colI[e];
            #pragma unroll
            for (int j = 0; j < 32; ++j) {
                float4 v = x4[(size_t)c * 32 + j];
                uint2 p; p.x = f2bf(v.x) | ((unsigned)f2bf(v.y) << 16);
                         p.y = f2bf(v.z) | ((unsigned)f2bf(v.w) << 16);
                *(uint2*)&st[lane * STRIDE_E + j * 4] = p;
            }
            #pragma unroll
            for (int j = 0; j < 4; ++j) {
                float4 v = ea4[(size_t)e * 4 + j];
                uint2 p; p.x = f2bf(v.x) | ((unsigned)f2bf(v.y) << 16);
                         p.y = f2bf(v.z) | ((unsigned)f2bf(v.w) << 16);
                *(uint2*)&st[lane * STRIDE_E + 128 + j * 4] = p;
            }
            uint2 z; z.x = 0u; z.y = 0u;
            #pragma unroll
            for (int j = 0; j < 4; ++j)
                *(uint2*)&st[lane * STRIDE_E + 144 + j * 4] = z;  // K pad 144..159
        }
        __syncthreads();

        // GEMM1: [32x160] x [160x128]
        v8f acc0[NT], acc1[NT];
        #pragma unroll
        for (int nt = 0; nt < NT; ++nt) { acc0[nt] = zv; acc1[nt] = zv; }
        #pragma unroll
        for (int ks = 0; ks < KS1; ++ks) {
            const u16* ap0 = &st[rA * STRIDE_E + ks * 32 + kh];
            const u16* ap1 = &st[(rA + 16) * STRIDE_E + ks * 32 + kh];
            v16bf a0 = make_frag(*(const v8us*)ap0, *(const v8us*)(ap0 + 16));
            v16bf a1 = make_frag(*(const v8us*)ap1, *(const v8us*)(ap1 + 16));
            v16bf bfr[NT];
            #pragma unroll
            for (int nt = 0; nt < NT; ++nt) {       // load all B frags first:
                const u16* bp = &sW1[((nt * KS1 + ks) * 32 + lane) * 16];
                bfr[nt] = make_frag(*(const v8us*)bp, *(const v8us*)(bp + 8));
            }
            #pragma unroll
            for (int nt = 0; nt < NT; ++nt) {       // then WMMAs back-to-back
                acc0[nt] = __builtin_amdgcn_wmma_f32_16x16x32_bf16(
                    false, a0, false, bfr[nt], (short)0, acc0[nt], false, false);
                acc1[nt] = __builtin_amdgcn_wmma_f32_16x16x32_bf16(
                    false, a1, false, bfr[nt], (short)0, acc1[nt], false, false);
            }
        }
        __syncthreads();

        // bias + ReLU, restage h1 (32x128) as bf16 in A layout
        #pragma unroll
        for (int nt = 0; nt < NT; ++nt) {
            const float bb = b1v[nt];
            #pragma unroll
            for (int j = 0; j < 8; ++j) {
                st[(ms + j) * STRIDE_E + nt * 16 + n0]      = f2bf(fmaxf(acc0[nt][j] + bb, 0.0f));
                st[(16 + ms + j) * STRIDE_E + nt * 16 + n0] = f2bf(fmaxf(acc1[nt][j] + bb, 0.0f));
            }
        }
        __syncthreads();

        // GEMM2: [32x128] x [128x128]
        #pragma unroll
        for (int nt = 0; nt < NT; ++nt) { acc0[nt] = zv; acc1[nt] = zv; }
        #pragma unroll
        for (int ks = 0; ks < KS2; ++ks) {
            const u16* ap0 = &st[rA * STRIDE_E + ks * 32 + kh];
            const u16* ap1 = &st[(rA + 16) * STRIDE_E + ks * 32 + kh];
            v16bf a0 = make_frag(*(const v8us*)ap0, *(const v8us*)(ap0 + 16));
            v16bf a1 = make_frag(*(const v8us*)ap1, *(const v8us*)(ap1 + 16));
            v16bf bfr[NT];
            #pragma unroll
            for (int nt = 0; nt < NT; ++nt) {
                const u16* bp = &sW2[((nt * KS2 + ks) * 32 + lane) * 16];
                bfr[nt] = make_frag(*(const v8us*)bp, *(const v8us*)(bp + 8));
            }
            #pragma unroll
            for (int nt = 0; nt < NT; ++nt) {
                acc0[nt] = __builtin_amdgcn_wmma_f32_16x16x32_bf16(
                    false, a0, false, bfr[nt], (short)0, acc0[nt], false, false);
                acc1[nt] = __builtin_amdgcn_wmma_f32_16x16x32_bf16(
                    false, a1, false, bfr[nt], (short)0, acc1[nt], false, false);
            }
        }

        if (valid) { // scatter-add into per-node accumulator (+ b2), counts
            #pragma unroll
            for (int nt = 0; nt < NT; ++nt) {
                const float bb = b2v[nt];
                #pragma unroll
                for (int j = 0; j < 8; ++j) {
                    const int e0 = base + ms + j;
                    const int e1 = base + 16 + ms + j;
                    if (e0 < E)
                        atomicAdd(&summed[(size_t)rowI[e0] * MID + nt * 16 + n0],
                                  acc0[nt][j] + bb);
                    if (e1 < E)
                        atomicAdd(&summed[(size_t)rowI[e1] * MID + nt * 16 + n0],
                                  acc1[nt][j] + bb);
                }
            }
            const int e = base + lane;
            if (e < E) atomicAdd(&cnt[rowI[e]], 1.0f);
        }
        __syncthreads();
    }
}

// ---------------------------------------------------------------- node MLP
__global__ __launch_bounds__(256)
void node_kernel(const float* __restrict__ summed, const float* __restrict__ cnt,
                 const float* __restrict__ b3, const float* __restrict__ b4,
                 const u16* __restrict__ gW3, const u16* __restrict__ gW4,
                 float* __restrict__ out, int N) {
    __shared__ alignas(16) u16 sW3[NT * KS2 * 32 * 16];
    __shared__ alignas(16) u16 sW4[NT * KS2 * 32 * 16];
    __shared__ alignas(16) u16 sStage[WAVES][16 * STRIDE_N];

    const int tid = threadIdx.x, w = tid >> 5, lane = tid & 31;
    {
        const uint4* g3 = (const uint4*)gW3; uint4* s3 = (uint4*)sW3;
        for (int i = tid; i < (NT * KS2 * 32 * 16) / 8; i += 256) s3[i] = g3[i];
        const uint4* g4 = (const uint4*)gW4; uint4* s4 = (uint4*)sW4;
        for (int i = tid; i < (NT * KS2 * 32 * 16) / 8; i += 256) s4[i] = g4[i];
    }
    __syncthreads();

    const float4* sm4 = (const float4*)summed;
    const int nTiles = (N + 15) / 16;
    const int tile0  = ((int)blockIdx.x * WAVES + w) * TPW_N;
    u16* st = &sStage[w][0];

    const int rA = lane & 15, kh = (lane >> 4) * 8;
    const int n0 = lane & 15, ms = (lane >> 4) * 8;
    const v8f zv = {0.f,0.f,0.f,0.f,0.f,0.f,0.f,0.f};

    float b3v[NT], b4v[NT];
    #pragma unroll
    for (int nt = 0; nt < NT; ++nt) { b3v[nt] = b3[nt * 16 + n0]; b4v[nt] = b4[nt * 16 + n0]; }

    for (int t = 0; t < TPW_N; ++t) {
        const int tile = tile0 + t;
        const int base = tile * 16;
        const bool valid = (tile < nTiles);

        if (valid) { // stage mean = summed / max(cnt,1) as bf16
            const int r = lane >> 1, half = lane & 1;
            int node = base + r; if (node >= N) node = N - 1;
            const float inv = 1.0f / fmaxf(cnt[node], 1.0f);
            #pragma unroll
            for (int j = 0; j < 16; ++j) {
                float4 v = sm4[(size_t)node * 32 + half * 16 + j];
                uint2 p; p.x = f2bf(v.x * inv) | ((unsigned)f2bf(v.y * inv) << 16);
                         p.y = f2bf(v.z * inv) | ((unsigned)f2bf(v.w * inv) << 16);
                *(uint2*)&st[r * STRIDE_N + half * 64 + j * 4] = p;
            }
        }
        __syncthreads();

        v8f acc[NT];
        #pragma unroll
        for (int nt = 0; nt < NT; ++nt) acc[nt] = zv;
        #pragma unroll
        for (int ks = 0; ks < KS2; ++ks) {
            const u16* ap = &st[rA * STRIDE_N + ks * 32 + kh];
            v16bf a = make_frag(*(const v8us*)ap, *(const v8us*)(ap + 16));
            v16bf bfr[NT];
            #pragma unroll
            for (int nt = 0; nt < NT; ++nt) {
                const u16* bp = &sW3[((nt * KS2 + ks) * 32 + lane) * 16];
                bfr[nt] = make_frag(*(const v8us*)bp, *(const v8us*)(bp + 8));
            }
            #pragma unroll
            for (int nt = 0; nt < NT; ++nt)
                acc[nt] = __builtin_amdgcn_wmma_f32_16x16x32_bf16(
                    false, a, false, bfr[nt], (short)0, acc[nt], false, false);
        }
        __syncthreads();

        #pragma unroll
        for (int nt = 0; nt < NT; ++nt) {
            const float bb = b3v[nt];
            #pragma unroll
            for (int j = 0; j < 8; ++j)
                st[(ms + j) * STRIDE_N + nt * 16 + n0] = f2bf(fmaxf(acc[nt][j] + bb, 0.0f));
        }
        __syncthreads();

        #pragma unroll
        for (int nt = 0; nt < NT; ++nt) acc[nt] = zv;
        #pragma unroll
        for (int ks = 0; ks < KS2; ++ks) {
            const u16* ap = &st[rA * STRIDE_N + ks * 32 + kh];
            v16bf a = make_frag(*(const v8us*)ap, *(const v8us*)(ap + 16));
            v16bf bfr[NT];
            #pragma unroll
            for (int nt = 0; nt < NT; ++nt) {
                const u16* bp = &sW4[((nt * KS2 + ks) * 32 + lane) * 16];
                bfr[nt] = make_frag(*(const v8us*)bp, *(const v8us*)(bp + 8));
            }
            #pragma unroll
            for (int nt = 0; nt < NT; ++nt)
                acc[nt] = __builtin_amdgcn_wmma_f32_16x16x32_bf16(
                    false, a, false, bfr[nt], (short)0, acc[nt], false, false);
        }

        if (valid) {
            #pragma unroll
            for (int nt = 0; nt < NT; ++nt) {
                const float bb = b4v[nt];
                #pragma unroll
                for (int j = 0; j < 8; ++j) {
                    const int node = base + ms + j;
                    if (node < N)
                        out[(size_t)node * MID + nt * 16 + n0] = acc[nt][j] + bb;
                }
            }
        }
        __syncthreads();
    }
}

// ---------------------------------------------------------------- launch
extern "C" void kernel_launch(void* const* d_in, const int* in_sizes, int n_in,
                              void* d_out, int out_size, void* d_ws, size_t ws_size,
                              hipStream_t stream) {
    const float* x    = (const float*)d_in[0];
    const int*   ei   = (const int*)  d_in[1];
    const float* ea   = (const float*)d_in[2];
    // d_in[3] = batch (unused)
    const float* W1 = (const float*)d_in[4];  const float* b1 = (const float*)d_in[5];
    const float* W2 = (const float*)d_in[6];  const float* b2 = (const float*)d_in[7];
    const float* W3 = (const float*)d_in[8];  const float* b3 = (const float*)d_in[9];
    const float* W4 = (const float*)d_in[10]; const float* b4 = (const float*)d_in[11];
    float* out = (float*)d_out;

    const int N = in_sizes[0] / 128;
    const int E = in_sizes[2] / 16;

    // workspace layout
    char* ws = (char*)d_ws;
    float* summed = (float*)ws;                 size_t o = (size_t)N * MID * sizeof(float);
    float* cntb   = (float*)(ws + o);           o += (size_t)N * sizeof(float);
    o = (o + 255) & ~(size_t)255;
    u16* wp1 = (u16*)(ws + o);                  o += (size_t)NT * KS1 * 512 * sizeof(u16);
    u16* wp2 = (u16*)(ws + o);                  o += (size_t)NT * KS2 * 512 * sizeof(u16);
    u16* wp3 = (u16*)(ws + o);                  o += (size_t)NT * KS2 * 512 * sizeof(u16);
    u16* wp4 = (u16*)(ws + o);                  o += (size_t)NT * KS2 * 512 * sizeof(u16);

    // 1) zero accumulators (every launch: atomics accumulate)
    {
        long n = (long)N * (MID + 1);
        zero_kernel<<<(unsigned)((n + 255) / 256), 256, 0, stream>>>(summed, n);
    }
    // 2) pack weights to bf16 WMMA B layout
    {
        int t1 = NT * KS1 * 512, t2 = NT * KS2 * 512;
        pack_w_kernel<<<(t1 + 255) / 256, 256, 0, stream>>>(W1, wp1, 144, KS1);
        pack_w_kernel<<<(t2 + 255) / 256, 256, 0, stream>>>(W2, wp2, 128, KS2);
        pack_w_kernel<<<(t2 + 255) / 256, 256, 0, stream>>>(W3, wp3, 128, KS2);
        pack_w_kernel<<<(t2 + 255) / 256, 256, 0, stream>>>(W4, wp4, 128, KS2);
    }
    // 3) edge MLP + scatter-add (32-edge tiles per wave)
    {
        int nTiles = (E + 31) / 32;
        int per_block = WAVES * TPW_E;
        int blocks = (nTiles + per_block - 1) / per_block;
        edge_kernel<<<blocks, 256, 0, stream>>>(x, ei, ea, b1, b2, wp1, wp2,
                                                summed, cntb, E);
    }
    // 4) node MLP
    {
        int nTiles = (N + 15) / 16;
        int per_block = WAVES * TPW_N;
        int blocks = (nTiles + per_block - 1) / per_block;
        node_kernel<<<blocks, 256, 0, stream>>>(summed, cntb, b3, b4, wp3, wp4,
                                                out, N);
    }
}
